// MetaLoss_32255204393276
// MI455X (gfx1250) — matrix operand order repaired and checked by reading, not compiled, not run on hardware
//
#include <hip/hip_runtime.h>
#include <hip/hip_bf16.h>

typedef __attribute__((ext_vector_type(16))) _Float16 v16h;
typedef __attribute__((ext_vector_type(8)))  _Float16 v8h;
typedef __attribute__((ext_vector_type(8)))  float    v8f;

#define BATCH   8
#define HDIM    192
#define WDIM    192
#define NIMG    (HDIM * WDIM)            // 36864
#define NPIX    (BATCH * NIMG)           // 294912
#define VALID   182                      // 192 - 11 + 1
#define NVALID  (BATCH * VALID * VALID)  // 264992

// ---------------- reduction helper (blockDim must be 256) ----------------
__device__ __forceinline__ float block_reduce_256(float v) {
    __shared__ float sm[256];
    int t = threadIdx.x;
    __syncthreads();              // protect shared from a previous call
    sm[t] = v;
    __syncthreads();
    #pragma unroll
    for (int s = 128; s > 0; s >>= 1) {
        if (t < s) sm[t] += sm[t + s];
        __syncthreads();
    }
    return sm[0];
}

// ---------------- init: zero accumulators, build banded Gaussian mats ----
// GhT is stored TRANSPOSED: GhT[n][k] = g[k-n]  (n < VALID, 0 <= k-n <= 10)
// GvT is stored row-major (A-operand of gemm2): GvT[m][y] = g[y-m]
__global__ void ml_init_kernel(_Float16* GhT, _Float16* GvT, float* acc, int* hasfg) {
    int i = blockIdx.x * 256 + threadIdx.x;
    float g[11]; float gs = 0.f;
    #pragma unroll
    for (int k = 0; k < 11; ++k) {
        float d = (float)k - 5.f;
        g[k] = __expf(-(d * d) / (2.f * 1.5f * 1.5f));
        gs += g[k];
    }
    #pragma unroll
    for (int k = 0; k < 11; ++k) g[k] /= gs;

    if (i < NIMG) {
        int r = i / WDIM, c = i % WDIM;
        // GhT[n=r][k=c]
        float v = 0.f;
        int d = c - r;
        if (r < VALID && d >= 0 && d <= 10) v = g[d];
        GhT[i] = (_Float16)v;
        // GvT[m=r][y=c]
        float v2 = 0.f;
        int d2 = c - r;
        if (r < VALID && d2 >= 0 && d2 <= 10) v2 = g[d2];
        GvT[i] = (_Float16)v2;
    }
    if (i < 8) acc[i] = 0.f;
    if (i >= 8 && i < 24) hasfg[i - 8] = 0;
}

// ---------------- prep: sigmoid, BCE, focal, f16 quantity maps ----------
__global__ void ml_prep_kernel(const float* __restrict__ pred,
                               const int* __restrict__ tgt,
                               float* __restrict__ p_f32,
                               _Float16* __restrict__ X,
                               float* acc) {
    int i = blockIdx.x * 256 + threadIdx.x;
    float bce = 0.f, foc = 0.f;
    if (i < NPIX) {
        float x = pred[i];
        float p = 1.f / (1.f + __expf(-x));
        float t = (tgt[i] != 0) ? 1.f : 0.f;
        p_f32[i] = p;
        X[(size_t)0 * NPIX + i] = (_Float16)p;
        X[(size_t)1 * NPIX + i] = (_Float16)t;
        X[(size_t)2 * NPIX + i] = (_Float16)(p * p);
        X[(size_t)3 * NPIX + i] = (_Float16)(t * t);
        X[(size_t)4 * NPIX + i] = (_Float16)(p * t);
        // BCE with log clamp at -100
        float lp = fmaxf(__logf(p), -100.f);
        float l1 = fmaxf(__logf(1.f - p), -100.f);
        bce = -(t * lp + (1.f - t) * l1);
        // focal
        float pc = fminf(fmaxf(p, 1e-6f), 1.f - 1e-6f);
        float pt = pc * t + (1.f - pc) * (1.f - t);
        float at = 0.25f * t + 0.75f * (1.f - t);
        float om = 1.f - pt;
        foc = -at * om * om * __logf(pt);
    }
    float s1 = block_reduce_256(bce);
    if (threadIdx.x == 0) atomicAdd(&acc[0], s1);
    float s2 = block_reduce_256(foc);
    if (threadIdx.x == 0) atomicAdd(&acc[1], s2);
}

// ---------------- EDT pass 1: per-row masked 1-D squared distance --------
__global__ void ml_colmin_kernel(const int* __restrict__ tgt,
                                 float* __restrict__ colmin,
                                 int* hasfg) {
    int r = blockIdx.x, b = blockIdx.y, pol = blockIdx.z;
    __shared__ float mask[WDIM];
    __shared__ int anyfg;
    int x = threadIdx.x;
    if (x == 0) anyfg = 0;
    __syncthreads();
    if (x < WDIM) {
        int fg = (tgt[((size_t)b * HDIM + r) * WDIM + x] != 0) ? 1 : 0;
        int m = (pol == 0) ? fg : (1 - fg);
        mask[x] = (float)m;
        if (m) anyfg = 1;      // benign race
    }
    __syncthreads();
    if (x < WDIM) {
        float best = 1e10f;
        for (int xp = 0; xp < WDIM; ++xp) {
            if (mask[xp] > 0.5f) {
                float d = (float)(x - xp);
                best = fminf(best, d * d);
            }
        }
        colmin[(((size_t)pol * BATCH + b) * HDIM + r) * WDIM + x] = best;
    }
    __syncthreads();
    if (x == 0 && anyfg) atomicOr(&hasfg[pol * BATCH + b], 1);
}

// ---------------- EDT pass 2 + boundary loss (fused) ---------------------
__global__ void ml_boundary_kernel(const float* __restrict__ pmap,
                                   const int* __restrict__ tgt,
                                   const float* __restrict__ colmin,
                                   const int* __restrict__ hasfg,
                                   float* acc) {
    int y = blockIdx.x, b = blockIdx.y;
    int x = threadIdx.x;
    float contrib = 0.f;
    if (x < WDIM) {
        float d2o = 1e10f, d2i = 1e10f;
        const float* cm0 = colmin + ((size_t)(0 * BATCH + b) * HDIM) * WDIM + x;
        const float* cm1 = colmin + ((size_t)(1 * BATCH + b) * HDIM) * WDIM + x;
        for (int r = 0; r < HDIM; ++r) {
            float dy = (float)(y - r);
            float dy2 = dy * dy;
            d2o = fminf(d2o, cm0[(size_t)r * WDIM] + dy2);
            d2i = fminf(d2i, cm1[(size_t)r * WDIM] + dy2);
        }
        const float maxd2 = 191.f * 191.f * 2.f;
        const float maxd  = sqrtf(maxd2);
        float dout = hasfg[b]         ? sqrtf(fminf(d2o, maxd2)) : maxd;
        float din  = hasfg[BATCH + b] ? sqrtf(fminf(d2i, maxd2)) : maxd;
        int fg = (tgt[((size_t)b * HDIM + y) * WDIM + x] != 0);
        float phi = fg ? -din : dout;
        float p = pmap[((size_t)b * HDIM + y) * WDIM + x];
        contrib = __expf(-fabsf(phi) * 0.1f) * fabsf(phi * p);
    }
    float s = block_reduce_256(contrib);
    if (threadIdx.x == 0) atomicAdd(&acc[2], s);
}

// ---------------- WMMA GEMM tiles (16x16 out, K in chunks of 32) ---------
// A-frag: lane l -> M = l&15, g = l>>4; elem i -> K = i + 16g
//         => two contiguous 8-half runs at k0+16g and k0+16g+8 (b128 loads)
// B-frag: lane l -> N = l&15, g = l>>4; elem i -> K = i + 16g
//         => one contiguous 16-half run of the TRANSPOSED B at k0+16g
// D-frag: lane l -> N = l&15, g = l>>4; vgpr v -> M = v + 8g
//         => 8 contiguous M values: transposed store is one b128
__device__ __forceinline__ v8f wmma_gemm_192(const _Float16* __restrict__ A,
                                             const _Float16* __restrict__ Bt,
                                             int m0, int n0) {
    int lane = threadIdx.x & 31;
    int g = lane >> 4;
    int mn = lane & 15;
    const _Float16* arow = A  + (size_t)(m0 + mn) * WDIM + 16 * g;
    const _Float16* brow = Bt + (size_t)(n0 + mn) * WDIM + 16 * g;
    v8f c = {};
    #pragma unroll
    for (int k0 = 0; k0 < 192; k0 += 32) {
        v8h alo = *(const v8h*)(arow + k0);
        v8h ahi = *(const v8h*)(arow + k0 + 8);
        v16h a = __builtin_shufflevector(alo, ahi, 0, 1, 2, 3, 4, 5, 6, 7,
                                         8, 9, 10, 11, 12, 13, 14, 15);
        v16h bb = *(const v16h*)(brow + k0);
        c = __builtin_amdgcn_wmma_f32_16x16x32_f16(false, a, false, bb,
                                                   (short)0, c, false, false);
    }
    return c;
}

// Tt = (X_qb * Gh)^T   (f16 out, stored transposed: Tt[n][m])
__global__ void ml_gemm_xg_kernel(const _Float16* __restrict__ X,
                                  const _Float16* __restrict__ GhT,
                                  _Float16* __restrict__ Tt) {
    int tile = blockIdx.x;                // 0..143 (12x12 tiles)
    int q = blockIdx.y, b = blockIdx.z;
    int m0 = (tile / 12) * 16, n0 = (tile % 12) * 16;
    const _Float16* A = X + (size_t)(q * BATCH + b) * NIMG;
    _Float16* D = Tt + (size_t)(q * BATCH + b) * NIMG;
    v8f c = wmma_gemm_192(A, GhT, m0, n0);
    int lane = threadIdx.x & 31;
    int g = lane >> 4, n = lane & 15;
    v8h dv;
    #pragma unroll
    for (int v = 0; v < 8; ++v) dv[v] = (_Float16)c[v];
    *(v8h*)(D + (size_t)(n0 + n) * WDIM + m0 + 8 * g) = dv;
}

// Ot = (GvT * T)^T  (f32 out, stored transposed: Ot[xo][yo])
// B operand is Tt (already transposed), so it is K-contiguous as needed.
__global__ void ml_gemm_gt_kernel(const _Float16* __restrict__ GvT,
                                  const _Float16* __restrict__ Tt,
                                  float* __restrict__ Ot) {
    int tile = blockIdx.x;
    int q = blockIdx.y, b = blockIdx.z;
    int m0 = (tile / 12) * 16, n0 = (tile % 12) * 16;
    const _Float16* Bm = Tt + (size_t)(q * BATCH + b) * NIMG;
    float* D = Ot + (size_t)(q * BATCH + b) * NIMG;
    v8f c = wmma_gemm_192(GvT, Bm, m0, n0);
    int lane = threadIdx.x & 31;
    int g = lane >> 4, n = lane & 15;
    *(v8f*)(D + (size_t)(n0 + n) * WDIM + m0 + 8 * g) = c;
}

// ---------------- SSIM combine over valid region -------------------------
// Ot layout: Ot[q][b][xo][yo]  (transposed spatial)
__global__ void ml_ssim_kernel(const float* __restrict__ Ot, float* acc) {
    int idx = blockIdx.x * 256 + threadIdx.x;
    float v = 0.f;
    if (idx < NVALID) {
        int b = idx / (VALID * VALID);
        int rem = idx % (VALID * VALID);
        int xo = rem / VALID, yo = rem % VALID;
        size_t base = (size_t)b * NIMG + (size_t)xo * WDIM + yo;
        size_t qs = (size_t)BATCH * NIMG;
        float mx  = Ot[0 * qs + base];
        float my  = Ot[1 * qs + base];
        float exx = Ot[2 * qs + base];
        float eyy = Ot[3 * qs + base];
        float exy = Ot[4 * qs + base];
        float sx = exx - mx * mx;
        float sy = eyy - my * my;
        float sxy = exy - mx * my;
        const float C1 = 1e-4f, C2 = 9e-4f;
        float num = (2.f * mx * my + C1) * (2.f * sxy + C2);
        float den = (mx * mx + my * my + C1) * (sx + sy + C2);
        v = num / den;
    }
    float s = block_reduce_256(v);
    if (threadIdx.x == 0) atomicAdd(&acc[3], s);
}

// ---------------- final scalar ------------------------------------------
__global__ void ml_final_kernel(const float* __restrict__ acc, float* out) {
    const float npix = (float)NPIX;
    const float nv = (float)NVALID;
    out[0] = acc[0] / npix + acc[1] / npix + acc[2] / npix + acc[3] / nv;
}

extern "C" void kernel_launch(void* const* d_in, const int* in_sizes, int n_in,
                              void* d_out, int out_size, void* d_ws, size_t ws_size,
                              hipStream_t stream) {
    (void)in_sizes; (void)n_in; (void)out_size;
    const float* pred = (const float*)d_in[0];
    const int*   tgt  = (const int*)d_in[1];
    float* out = (float*)d_out;
    char* ws = (char*)d_ws;

    // workspace carve-up (all 256B aligned)
    size_t off = 0;
    float* acc   = (float*)(ws + off);                    // 8 floats
    int*   hasfg = (int*)(ws + 32);                       // 16 ints
    off = 256;
    float*    p_f32  = (float*)(ws + off);    off += (size_t)NPIX * 4;        // 1,179,648
    _Float16* X      = (_Float16*)(ws + off); off += (size_t)5 * NPIX * 2;    // 2,949,120
    _Float16* GhT    = (_Float16*)(ws + off); off += (size_t)NIMG * 2;        //    73,728
    _Float16* GvT    = (_Float16*)(ws + off); off += (size_t)NIMG * 2;        //    73,728
    _Float16* Tt     = (_Float16*)(ws + off); off += (size_t)5 * NPIX * 2;    // 2,949,120
    float*    Ot     = (float*)(ws + off);    off += (size_t)5 * NPIX * 4;    // 5,898,240
    float*    colmin = (float*)(ws + off);    off += (size_t)2 * NPIX * 4;    // 2,359,296
    if (ws_size < off) return;  // not enough scratch; cannot compute

    // 1) init: zero accumulators, build Gaussian band matrices
    ml_init_kernel<<<(NIMG + 255) / 256, 256, 0, stream>>>(GhT, GvT, acc, hasfg);
    // 2) sigmoid + BCE + focal + f16 quantity maps
    ml_prep_kernel<<<(NPIX + 255) / 256, 256, 0, stream>>>(pred, tgt, p_f32, X, acc);
    // 3) EDT row pass (both polarities)
    ml_colmin_kernel<<<dim3(HDIM, BATCH, 2), 256, 0, stream>>>(tgt, colmin, hasfg);
    // 4) EDT column pass fused with boundary loss
    ml_boundary_kernel<<<dim3(HDIM, BATCH), 256, 0, stream>>>(p_f32, tgt, colmin, hasfg, acc);
    // 5) SSIM horizontal conv as WMMA GEMM: Tt = (X * Gh)^T
    ml_gemm_xg_kernel<<<dim3(144, 5, BATCH), 32, 0, stream>>>(X, GhT, Tt);
    // 6) SSIM vertical conv as WMMA GEMM: Ot = (GvT * T)^T
    ml_gemm_gt_kernel<<<dim3(144, 5, BATCH), 32, 0, stream>>>(GvT, Tt, Ot);
    // 7) SSIM per-pixel combine + reduce
    ml_ssim_kernel<<<(NVALID + 255) / 256, 256, 0, stream>>>(Ot, acc);
    // 8) final scalar
    ml_final_kernel<<<1, 1, 0, stream>>>(acc, out);
}
